// AttentionBlocks_59940563583777
// MI455X (gfx1250) — matrix operand order repaired and checked by reading, not compiled, not run on hardware
//
#include <hip/hip_runtime.h>
#include <cstdint>
#include <cstddef>

// ---------------------------------------------------------------------------
// MI455X (gfx1250) implementation of a 3D shifted-window attention block.
// GEMMs: v_wmma_f32_16x16x32_f16 (wave32 WMMA, f16 operands, f32 acc),
// 16x64 output strip per wave (A-fragment reused across 4 B tiles).
// Attention stages its P@V B-operand into LDS with
// global_load_async_to_lds_b128 (ASYNCcnt), overlapped with S + softmax.
// ---------------------------------------------------------------------------

typedef __attribute__((ext_vector_type(16))) _Float16 v16h;
typedef __attribute__((ext_vector_type(8)))  float    v8f;

#define DEV __device__ __forceinline__

// ---- problem dimensions ----------------------------------------------------
constexpr int BB   = 2, TT = 4, HH = 70, WW = 70, CC = 180;
constexpr int HEADS = 6, HD = 30, HDP = 32;
constexpr int NTOK = BB * TT * HH * WW;      // 39200 tokens
constexpr int LDK  = 192;                    // K=180 padded to 192 (6 x 32)
constexpr int NWIN = 400;                    // 2 batches * 200 windows
constexpr int NMASK = 200;                   // distinct mask windows
constexpr int NT = 98, NTP = 112, KT = 128;  // tokens/window (98 -> 112 / 128)
constexpr int MLP = 720, MLPP = 736;         // 4*DIM, padded to 23*32
constexpr float QSCALE = 0.18257418583505536f; // 30^-0.5

// ---- WMMA helpers ----------------------------------------------------------
DEV v8f wmma16(v16h a, v16h b, v8f c) {
  return __builtin_amdgcn_wmma_f32_16x16x32_f16(
      /*neg_a=*/false, a, /*neg_b=*/false, b,
      /*c_mod=*/(short)0, c, /*reuse_a=*/false, /*reuse_b=*/false);
}

// A fragment: 16x32 f16 tile, row-major, leading dim `ldk` (halves).
// element e of lane L holds A[m][k], m = L&15,
// k = (e&7) + ((L&16)?8:0) + ((e&8)?16:0)   (ISA 7.12.2, 16-bit A 16x32)
DEV v16h load_a_frag(const _Float16* tile, int ldk, int lane) {
  const int m  = lane & 15;
  const int kh = (lane & 16) ? 8 : 0;
  const unsigned* p = (const unsigned*)(tile + (size_t)m * ldk + kh);
  union { v16h v; unsigned u[8]; } A;
#pragma unroll
  for (int t = 0; t < 4; ++t) { A.u[t] = p[t]; A.u[4 + t] = p[8 + t]; }
  return A.v;
}

// B fragment: B is 32x16 (K x N), stored transposed as [n][k].
// element e of lane L holds B[k][n], n = L&15,
// k = (e&7) + ((L&16)?8:0) + ((e&8)?16:0).
DEV v16h load_b_frag(const _Float16* tileNK, int ldk, int lane) {
  const int n  = lane & 15;
  const int kh = (lane & 16) ? 8 : 0;
  const unsigned* p = (const unsigned*)(tileNK + (size_t)n * ldk + kh);
  union { v16h v; unsigned u[8]; } Bf;
#pragma unroll
  for (int t = 0; t < 4; ++t) { Bf.u[t] = p[t]; Bf.u[4 + t] = p[8 + t]; }
  return Bf.v;
}

// C/D fragment mapping: acc[r] = D[m][n], m = r + 8*(lane>>4), n = lane&15.

DEV float wredsum32(float v) {
#pragma unroll
  for (int m = 16; m; m >>= 1) v += __shfl_xor(v, m, 32);
  return v;
}
DEV float wredsum16(float v) {
#pragma unroll
  for (int m = 8; m; m >>= 1) v += __shfl_xor(v, m, 32);
  return v;
}
DEV float wredmax16(float v) {
#pragma unroll
  for (int m = 8; m; m >>= 1) v = fmaxf(v, __shfl_xor(v, m, 32));
  return v;
}

DEV float gelu_exact(float v) {
  return 0.5f * v * (1.0f + erff(v * 0.70710678118654752f));
}

// ---------------------------------------------------------------------------
// LayerNorm over C=180.  One wave per token.  Optional f32 out and/or
// f16 out padded to LDK=192 (pad columns zeroed).
// ---------------------------------------------------------------------------
__global__ void k_ln(const float* __restrict__ src, const float* __restrict__ g,
                     const float* __restrict__ bta, float* __restrict__ dst32,
                     _Float16* __restrict__ dst16) {
  const int wid  = (blockIdx.x * blockDim.x + threadIdx.x) >> 5;
  const int lane = threadIdx.x & 31;
  if (wid >= NTOK) return;
  const float* p = src + (size_t)wid * CC;
  float v[6];
  float s = 0.f;
#pragma unroll
  for (int j = 0; j < 6; ++j) {
    const int c = lane + 32 * j;
    v[j] = (c < CC) ? p[c] : 0.f;
    s += v[j];
  }
  s = wredsum32(s);
  const float mean = s * (1.f / CC);
  float var = 0.f;
#pragma unroll
  for (int j = 0; j < 6; ++j) {
    const int c = lane + 32 * j;
    const float d = (c < CC) ? (v[j] - mean) : 0.f;
    var += d * d;
  }
  var = wredsum32(var) * (1.f / CC);
  const float rs = rsqrtf(var + 1e-5f);
#pragma unroll
  for (int j = 0; j < 6; ++j) {
    const int c = lane + 32 * j;
    if (c < CC) {
      const float o = (v[j] - mean) * rs * g[c] + bta[c];
      if (dst32) dst32[(size_t)wid * CC + c] = o;
      if (dst16) dst16[(size_t)wid * LDK + c] = (_Float16)o;
    } else if (dst16) {
      dst16[(size_t)wid * LDK + c] = (_Float16)0.f;
    }
  }
}

// ---------------------------------------------------------------------------
// Pack an f32 [rows x K] weight matrix into zero-padded f16 [dstRows x ldk].
// ---------------------------------------------------------------------------
__global__ void k_pack_w(const float* __restrict__ w, _Float16* __restrict__ dst,
                         int rows, int K, int ldk, int dstRows) {
  const size_t n = (size_t)dstRows * ldk;
  size_t i = (size_t)blockIdx.x * blockDim.x + threadIdx.x;
  if (i >= n) return;
  const int r = (int)(i / ldk), c = (int)(i % ldk);
  dst[i] = (r < rows && c < K) ? (_Float16)w[(size_t)r * K + c] : (_Float16)0.f;
}

__global__ void k_fill_h(_Float16* __restrict__ p, unsigned long long n) {
  size_t i  = (size_t)blockIdx.x * blockDim.x + threadIdx.x;
  size_t st = (size_t)gridDim.x * blockDim.x;
  for (; i < n; i += st) p[i] = (_Float16)0.f;
}

// ---------------------------------------------------------------------------
// Gather LN output into shifted-window row order, f16, K padded to 192.
// sx[t'] = xn[(t'+1)%4, (h'+3)%70, (w'+3)%70].
// ---------------------------------------------------------------------------
__global__ void k_pack_xwin(const float* __restrict__ xn, _Float16* __restrict__ xwin) {
  const size_t total = (size_t)NTOK * LDK;
  size_t i = (size_t)blockIdx.x * blockDim.x + threadIdx.x;
  if (i >= total) return;
  const int c = (int)(i % LDK);
  const int row = (int)(i / LDK);
  if (c >= CC) { xwin[i] = (_Float16)0.f; return; }
  const int win = row / NT, n = row % NT;
  const int b  = win / 200;
  const int tw = (win / 100) % 2;
  const int hw = (win / 10) % 10;
  const int ww = win % 10;
  const int nt = n / 49, nh = (n / 7) % 7, nw = n % 7;
  int t = tw * 2 + nt + 1; t &= 3;
  int h = hw * 7 + nh + 3; if (h >= HH) h -= HH;
  int w = ww * 7 + nw + 3; if (w >= WW) w -= WW;
  const size_t src = ((((size_t)b * TT + t) * HH + h) * WW + w) * CC + c;
  xwin[i] = (_Float16)xn[src];
}

// ---------------------------------------------------------------------------
// QKV GEMM: 16x64 strip per wave.  [39200 x 192] @ [576 x 192] -> scatter
// into q[win][head][112][32] (pre-scaled), k[...], vT[wh][32][128].
// ---------------------------------------------------------------------------
__global__ void k_gemm_qkv(const _Float16* __restrict__ xwin,
                           const _Float16* __restrict__ wq,
                           const float* __restrict__ qkv_b,
                           _Float16* __restrict__ qb, _Float16* __restrict__ kb,
                           _Float16* __restrict__ vT) {
  const int rt = blockIdx.x, cg = blockIdx.y, lane = threadIdx.x;
  v8f acc[4];
#pragma unroll
  for (int s2 = 0; s2 < 4; ++s2) acc[s2] = (v8f){};
#pragma unroll
  for (int kc = 0; kc < 6; ++kc) {
    if (kc < 5)
      __builtin_prefetch(xwin + ((size_t)rt * 16) * LDK + (kc + 1) * 32, 0, 3);
    v16h a = load_a_frag(xwin + ((size_t)rt * 16) * LDK + kc * 32, LDK, lane);
#pragma unroll
    for (int s2 = 0; s2 < 4; ++s2) {
      v16h b = load_b_frag(wq + ((size_t)(cg * 64 + s2 * 16)) * LDK + kc * 32, LDK, lane);
      acc[s2] = wmma16(a, b, acc[s2]);
    }
  }
  const int hi = (lane >> 4) & 1;
#pragma unroll
  for (int s2 = 0; s2 < 4; ++s2) {
    const int col = cg * 64 + s2 * 16 + (lane & 15);
    if (col >= 3 * CC) continue;
    const float bias = qkv_b[col];
    const int which = col / CC, cc = col % CC, head = cc / HD, hd = cc % HD;
#pragma unroll
    for (int r = 0; r < 8; ++r) {
      const int row = rt * 16 + r + hi * 8;
      const int win = row / NT, n = row % NT;
      const float v = acc[s2][r] + bias;
      const size_t qkIdx = (((size_t)win * HEADS + head) * NTP + n) * HDP + hd;
      if (which == 0)      qb[qkIdx] = (_Float16)(v * QSCALE);
      else if (which == 1) kb[qkIdx] = (_Float16)v;
      else vT[(((size_t)win * HEADS + head) * HDP + hd) * KT + n] = (_Float16)v;
    }
  }
}

// ---------------------------------------------------------------------------
// Windowed attention: one wave per (window, head, 16-row tile).
// vT (8KB) is staged to LDS via global_load_async_to_lds_b128, overlapped
// with S = q@k^T (7 WMMAs), bias+mask, and row softmax; s_wait_asynccnt 0
// fences before the P@V WMMAs.
// ---------------------------------------------------------------------------
__global__ void k_attn(const _Float16* __restrict__ qb,
                       const _Float16* __restrict__ kb,
                       const _Float16* __restrict__ vT,
                       const int* __restrict__ rpi, const float* __restrict__ rpb,
                       const float* __restrict__ mask,
                       _Float16* __restrict__ aout) {
  __shared__ __align__(16) _Float16 lp[16 * KT];    // P staging (4KB)
  __shared__ __align__(16) _Float16 vtl[HDP * KT];  // vT tile    (8KB)
  const int rt = blockIdx.x, head = blockIdx.y, win = blockIdx.z;
  const int lane = threadIdx.x;
  const int wh = win * HEADS + head;
  const int hi = (lane >> 4) & 1, colL = lane & 15;
  const int mw = win % NMASK;

  // ---- kick off async copy of vT[wh] (32x128 f16 = 8KB) into LDS ----------
  {
    // generic shared pointer low 32 bits == wave-relative LDS byte offset
    const unsigned ldsbase = (unsigned)(uintptr_t)(&vtl[0]);
    const unsigned long long gbase =
        (unsigned long long)(uintptr_t)(vT + (size_t)wh * HDP * KT);
#pragma unroll
    for (int it = 0; it < 16; ++it) {
      const unsigned boff = (unsigned)((it * 32 + lane) * 16);
      const unsigned lo = ldsbase + boff;
      const unsigned long long ga = gbase + boff;
      asm volatile("global_load_async_to_lds_b128 %0, %1, off"
                   :: "v"(lo), "v"(ga) : "memory");
    }
  }

  // ---- S = q @ k^T ---------------------------------------------------------
  const v16h aq = load_a_frag(qb + ((size_t)wh * NTP + rt * 16) * HDP, HDP, lane);
  v8f s[7];
#pragma unroll
  for (int j = 0; j < 7; ++j) {
    v16h bk = load_b_frag(kb + ((size_t)wh * NTP + j * 16) * HDP, HDP, lane);
    v8f z = {};
    s[j] = wmma16(aq, bk, z);
  }
  // bias + mask
#pragma unroll
  for (int j = 0; j < 7; ++j) {
    const int nk = j * 16 + colL;
#pragma unroll
    for (int r = 0; r < 8; ++r) {
      const int nq = rt * 16 + r + hi * 8;
      float v = s[j][r];
      if (nk < NT && nq < NT) {
        v += rpb[(size_t)rpi[nq * NT + nk] * HEADS + head];
        v += mask[((size_t)mw * NT + nq) * NT + nk];
      } else {
        v = -1e30f;
      }
      s[j][r] = v;
    }
  }
  // row softmax (each row lives in one 16-lane half: xor 1,2,4,8)
#pragma unroll
  for (int r = 0; r < 8; ++r) {
    float mx = -1e30f;
#pragma unroll
    for (int j = 0; j < 7; ++j) mx = fmaxf(mx, s[j][r]);
    mx = wredmax16(mx);
    float sum = 0.f;
#pragma unroll
    for (int j = 0; j < 7; ++j) {
      const float e = __expf(s[j][r] - mx);
      s[j][r] = e;
      sum += e;
    }
    sum = wredsum16(sum);
    const float inv = 1.f / sum;
#pragma unroll
    for (int j = 0; j < 7; ++j) s[j][r] *= inv;
  }
  // stage P into LDS as f16 [16][128] (cols 112..127 zero)
#pragma unroll
  for (int j = 0; j < 7; ++j)
#pragma unroll
    for (int r = 0; r < 8; ++r)
      lp[(r + hi * 8) * KT + j * 16 + colL] = (_Float16)s[j][r];
#pragma unroll
  for (int r = 0; r < 8; ++r)
    lp[(r + hi * 8) * KT + 112 + colL] = (_Float16)0.f;
  __syncthreads();

  // ---- fence the async vT copy, then out = P @ vT --------------------------
  asm volatile("s_wait_asynccnt 0x0" ::: "memory");

  v8f o0 = {}, o1 = {};
#pragma unroll
  for (int kc = 0; kc < 4; ++kc) {
    v16h ap = load_a_frag(lp + kc * 32, KT, lane);
    v16h b0 = load_b_frag(vtl + (size_t)(0)  * KT + kc * 32, KT, lane);
    v16h b1 = load_b_frag(vtl + (size_t)(16) * KT + kc * 32, KT, lane);
    o0 = wmma16(ap, b0, o0);
    o1 = wmma16(ap, b1, o1);
  }
#pragma unroll
  for (int r = 0; r < 8; ++r) {
    const int nq = rt * 16 + r + hi * 8;
    if (nq >= NT) continue;
    const int n0 = colL;        // from o0
    const int n1 = 16 + colL;   // from o1
    const size_t rowBase = ((size_t)win * NT + nq) * LDK + head * HD;
    if (n0 < HD) aout[rowBase + n0] = (_Float16)o0[r];
    if (n1 < HD) aout[rowBase + n1] = (_Float16)o1[r];
  }
}

// ---------------------------------------------------------------------------
// proj GEMM: 16x64 strip per wave; scatter (window_reverse + roll) -> xa f32.
// ---------------------------------------------------------------------------
__global__ void k_gemm_proj(const _Float16* __restrict__ aout,
                            const _Float16* __restrict__ wp,
                            const float* __restrict__ pb,
                            float* __restrict__ xa) {
  const int rt = blockIdx.x, cg = blockIdx.y, lane = threadIdx.x;
  v8f acc[4];
#pragma unroll
  for (int s2 = 0; s2 < 4; ++s2) acc[s2] = (v8f){};
#pragma unroll
  for (int kc = 0; kc < 6; ++kc) {
    v16h a = load_a_frag(aout + ((size_t)rt * 16) * LDK + kc * 32, LDK, lane);
#pragma unroll
    for (int s2 = 0; s2 < 4; ++s2) {
      v16h b = load_b_frag(wp + ((size_t)(cg * 64 + s2 * 16)) * LDK + kc * 32, LDK, lane);
      acc[s2] = wmma16(a, b, acc[s2]);
    }
  }
  const int hi = (lane >> 4) & 1;
#pragma unroll
  for (int s2 = 0; s2 < 4; ++s2) {
    const int col = cg * 64 + s2 * 16 + (lane & 15);
    if (col >= CC) continue;
    const float bias = pb[col];
#pragma unroll
    for (int r = 0; r < 8; ++r) {
      const int row = rt * 16 + r + hi * 8;
      const int win = row / NT, n = row % NT;
      const int b  = win / 200;
      const int tw = (win / 100) % 2;
      const int hw = (win / 10) % 10;
      const int ww = win % 10;
      const int nt = n / 49, nh = (n / 7) % 7, nw = n % 7;
      int t = tw * 2 + nt + 1; t &= 3;
      int h = hw * 7 + nh + 3; if (h >= HH) h -= HH;
      int w = ww * 7 + nw + 3; if (w >= WW) w -= WW;
      xa[((((size_t)b * TT + t) * HH + h) * WW + w) * CC + col] = acc[s2][r] + bias;
    }
  }
}

// ---------------------------------------------------------------------------
// Conv branch (NDHWC layout on xn): conv1 1x3x3 (180->60) + exact GELU.
// ---------------------------------------------------------------------------
__global__ void k_conv1(const float* __restrict__ xn, const float* __restrict__ w1,
                        const float* __restrict__ b1c, float* __restrict__ t1) {
  const size_t total = (size_t)NTOK * 60;
  size_t i = (size_t)blockIdx.x * blockDim.x + threadIdx.x;
  if (i >= total) return;
  const int oc = (int)(i % 60);
  size_t sp = i / 60;
  const int w_ = (int)(sp % WW); sp /= WW;
  const int h_ = (int)(sp % HH); sp /= HH;
  const int t_ = (int)(sp % TT);
  const int b_ = (int)(sp / TT);
  float acc = b1c[oc];
#pragma unroll
  for (int dh = -1; dh <= 1; ++dh) {
    const int hh = h_ + dh;
    if (hh < 0 || hh >= HH) continue;
#pragma unroll
    for (int dw = -1; dw <= 1; ++dw) {
      const int ww2 = w_ + dw;
      if (ww2 < 0 || ww2 >= WW) continue;
      const float* xp = xn + ((((size_t)b_ * TT + t_) * HH + hh) * WW + ww2) * CC;
      const float* wp = w1 + ((size_t)oc * CC) * 9 + (size_t)((dh + 1) * 3 + (dw + 1));
      for (int ic = 0; ic < CC; ++ic) acc += xp[ic] * wp[(size_t)ic * 9];
    }
  }
  t1[i] = gelu_exact(acc);
}

// conv2 1x3x3 (60->180), no activation.
__global__ void k_conv2(const float* __restrict__ t1, const float* __restrict__ w2,
                        const float* __restrict__ b2c, float* __restrict__ y) {
  const size_t total = (size_t)NTOK * CC;
  size_t i = (size_t)blockIdx.x * blockDim.x + threadIdx.x;
  if (i >= total) return;
  const int oc = (int)(i % CC);
  size_t sp = i / CC;
  const int w_ = (int)(sp % WW); sp /= WW;
  const int h_ = (int)(sp % HH); sp /= HH;
  const int t_ = (int)(sp % TT);
  const int b_ = (int)(sp / TT);
  float acc = b2c[oc];
#pragma unroll
  for (int dh = -1; dh <= 1; ++dh) {
    const int hh = h_ + dh;
    if (hh < 0 || hh >= HH) continue;
#pragma unroll
    for (int dw = -1; dw <= 1; ++dw) {
      const int ww2 = w_ + dw;
      if (ww2 < 0 || ww2 >= WW) continue;
      const float* tp = t1 + ((((size_t)b_ * TT + t_) * HH + hh) * WW + ww2) * 60;
      const float* wp = w2 + ((size_t)oc * 60) * 9 + (size_t)((dh + 1) * 3 + (dw + 1));
      for (int ic = 0; ic < 60; ++ic) acc += tp[ic] * wp[(size_t)ic * 9];
    }
  }
  y[i] = acc;
}

// Global average pool over (T,H,W) per (b,c).
__global__ void k_pool(const float* __restrict__ y, float* __restrict__ pool) {
  __shared__ float red[256];
  const int bc = blockIdx.x;          // 0..359
  const int b = bc / CC, c = bc % CC;
  const int per = TT * HH * WW;       // 19600
  float s = 0.f;
  for (int i = threadIdx.x; i < per; i += blockDim.x)
    s += y[((size_t)b * per + i) * CC + c];
  red[threadIdx.x] = s;
  __syncthreads();
  for (int st = 128; st; st >>= 1) {
    if (threadIdx.x < st) red[threadIdx.x] += red[threadIdx.x + st];
    __syncthreads();
  }
  if (threadIdx.x == 0) pool[bc] = red[0] * (1.f / per);
}

// Channel attention squeeze-excite: 180 -> 6 (relu) -> 180 (sigmoid).
__global__ void k_ca(const float* __restrict__ pool,
                     const float* __restrict__ a1w, const float* __restrict__ a1b,
                     const float* __restrict__ a2w, const float* __restrict__ a2b,
                     float* __restrict__ asig) {
  __shared__ float hid[BB * 6];
  const int t = threadIdx.x;
  if (t < BB * 6) {
    const int b = t / 6, j = t % 6;
    float s = a1b[j];
    for (int c = 0; c < CC; ++c) s += pool[b * CC + c] * a1w[(size_t)j * CC + c];
    hid[t] = fmaxf(s, 0.f);
  }
  __syncthreads();
  if (t < BB * CC) {
    const int b = t / CC, c = t % CC;
    float s = a2b[c];
#pragma unroll
    for (int j = 0; j < 6; ++j) s += hid[b * 6 + j] * a2w[(size_t)c * 6 + j];
    asig[t] = 1.f / (1.f + __expf(-s));
  }
}

// x = shortcut + attn_x + 0.01 * (y * channel_attention)
__global__ void k_combine(const float* __restrict__ x, const float* __restrict__ xa,
                          const float* __restrict__ y, const float* __restrict__ asig,
                          float* __restrict__ out) {
  const size_t total = (size_t)NTOK * CC;
  size_t i = (size_t)blockIdx.x * blockDim.x + threadIdx.x;
  if (i >= total) return;
  const int c = (int)(i % CC);
  const int b = (int)(i / ((size_t)TT * HH * WW * CC));
  out[i] = x[i] + xa[i] + 0.01f * y[i] * asig[b * CC + c];
}

// fc1 GEMM: 16x64 strip per wave; [39200 x 192] @ [768 x 192] -> gelu -> hbuf.
__global__ void k_gemm_fc1(const _Float16* __restrict__ xm,
                           const _Float16* __restrict__ w1,
                           const float* __restrict__ b1v,
                           _Float16* __restrict__ hbuf) {
  const int rt = blockIdx.x, cg = blockIdx.y, lane = threadIdx.x;
  v8f acc[4];
#pragma unroll
  for (int s2 = 0; s2 < 4; ++s2) acc[s2] = (v8f){};
#pragma unroll
  for (int kc = 0; kc < 6; ++kc) {
    v16h a = load_a_frag(xm + ((size_t)rt * 16) * LDK + kc * 32, LDK, lane);
#pragma unroll
    for (int s2 = 0; s2 < 4; ++s2) {
      v16h b = load_b_frag(w1 + ((size_t)(cg * 64 + s2 * 16)) * LDK + kc * 32, LDK, lane);
      acc[s2] = wmma16(a, b, acc[s2]);
    }
  }
  const int hi = (lane >> 4) & 1;
#pragma unroll
  for (int s2 = 0; s2 < 4; ++s2) {
    const int col = cg * 64 + s2 * 16 + (lane & 15);
    if (col >= MLP) continue;
    const float bias = b1v[col];
#pragma unroll
    for (int r = 0; r < 8; ++r) {
      const int row = rt * 16 + r + hi * 8;
      hbuf[(size_t)row * MLPP + col] = (_Float16)gelu_exact(acc[s2][r] + bias);
    }
  }
}

// fc2 GEMM: 16x64 strip per wave; [39200 x 736] @ [192 x 736] -> += d_out.
__global__ void k_gemm_fc2(const _Float16* __restrict__ hbuf,
                           const _Float16* __restrict__ w2,
                           const float* __restrict__ b2v,
                           float* __restrict__ out) {
  const int rt = blockIdx.x, cg = blockIdx.y, lane = threadIdx.x;
  v8f acc[4];
#pragma unroll
  for (int s2 = 0; s2 < 4; ++s2) acc[s2] = (v8f){};
  for (int kc = 0; kc < 23; ++kc) {
    v16h a = load_a_frag(hbuf + ((size_t)rt * 16) * MLPP + kc * 32, MLPP, lane);
#pragma unroll
    for (int s2 = 0; s2 < 4; ++s2) {
      v16h b = load_b_frag(w2 + ((size_t)(cg * 64 + s2 * 16)) * MLPP + kc * 32, MLPP, lane);
      acc[s2] = wmma16(a, b, acc[s2]);
    }
  }
  const int hi = (lane >> 4) & 1;
#pragma unroll
  for (int s2 = 0; s2 < 4; ++s2) {
    const int col = cg * 64 + s2 * 16 + (lane & 15);
    if (col >= CC) continue;
    const float bias = b2v[col];
#pragma unroll
    for (int r = 0; r < 8; ++r) {
      const int row = rt * 16 + r + hi * 8;
      const size_t o = (size_t)row * CC + col;
      out[o] = out[o] + acc[s2][r] + bias;
    }
  }
}

// ---------------------------------------------------------------------------
extern "C" void kernel_launch(void* const* d_in, const int* in_sizes, int n_in,
                              void* d_out, int out_size, void* d_ws, size_t ws_size,
                              hipStream_t stream) {
  (void)in_sizes; (void)n_in; (void)out_size; (void)ws_size;
  const float* x       = (const float*)d_in[0];
  const int*   rpi     = (const int*)d_in[1];
  const float* amask   = (const float*)d_in[2];
  const float* g1      = (const float*)d_in[3];
  const float* b1      = (const float*)d_in[4];
  const float* qkv_w   = (const float*)d_in[5];
  const float* qkv_b   = (const float*)d_in[6];
  const float* rpb     = (const float*)d_in[7];
  const float* proj_w  = (const float*)d_in[8];
  const float* proj_b  = (const float*)d_in[9];
  const float* conv1_w = (const float*)d_in[10];
  const float* conv1_b = (const float*)d_in[11];
  const float* conv2_w = (const float*)d_in[12];
  const float* conv2_b = (const float*)d_in[13];
  const float* ca1_w   = (const float*)d_in[14];
  const float* ca1_b   = (const float*)d_in[15];
  const float* ca2_w   = (const float*)d_in[16];
  const float* ca2_b   = (const float*)d_in[17];
  const float* g2      = (const float*)d_in[18];
  const float* b2      = (const float*)d_in[19];
  const float* fc1_w   = (const float*)d_in[20];
  const float* fc1_b   = (const float*)d_in[21];
  const float* fc2_w   = (const float*)d_in[22];
  const float* fc2_b   = (const float*)d_in[23];
  float* out = (float*)d_out;

  // -- workspace arena (deterministic offsets, 256B aligned) --
  char* base = (char*)d_ws;
  size_t off = 0;
  auto take = [&](size_t bytes) -> void* {
    void* p = base + off;
    off = (off + bytes + 255) & ~(size_t)255;
    return p;
  };
  float*    xn    = (float*)   take((size_t)NTOK * CC * 4);
  _Float16* xwin  = (_Float16*)take((size_t)NTOK * LDK * 2);
  _Float16* wq    = (_Float16*)take((size_t)576 * LDK * 2);
  _Float16* wproj = (_Float16*)take((size_t)192 * LDK * 2);
  _Float16* wfc1  = (_Float16*)take((size_t)768 * LDK * 2);
  _Float16* wfc2  = (_Float16*)take((size_t)192 * MLPP * 2);
  const size_t QKN = (size_t)NWIN * HEADS * NTP * HDP;   // 8,601,600
  const size_t VTN = (size_t)NWIN * HEADS * HDP * KT;    // 9,830,400
  _Float16* qb    = (_Float16*)take(QKN * 2);
  _Float16* kb    = (_Float16*)take(QKN * 2);
  _Float16* vT    = (_Float16*)take(VTN * 2);
  _Float16* aoutb = (_Float16*)take((size_t)NTOK * LDK * 2);
  float*    xa    = (float*)   take((size_t)NTOK * CC * 4);
  float*    t1    = (float*)   take((size_t)NTOK * 60 * 4);
  float*    ybuf  = (float*)   take((size_t)NTOK * CC * 4);
  float*    pool  = (float*)   take((size_t)BB * CC * 4);
  float*    asig  = (float*)   take((size_t)BB * CC * 4);
  _Float16* xm    = (_Float16*)take((size_t)NTOK * LDK * 2);
  _Float16* hbuf  = (_Float16*)take((size_t)NTOK * MLPP * 2);

  const int RT = NTOK / 16;   // 2450 row tiles

  // 1) LN1 -> xn (f32)
  k_ln<<<NTOK / 8, 256, 0, stream>>>(x, g1, b1, xn, nullptr);

  // 2) pack weights to padded f16 [n][k]
  auto packGrid = [](size_t n) { return dim3((unsigned)((n + 255) / 256)); };
  k_pack_w<<<packGrid((size_t)576 * LDK), 256, 0, stream>>>(qkv_w, wq, 3 * CC, CC, LDK, 576);
  k_pack_w<<<packGrid((size_t)192 * LDK), 256, 0, stream>>>(proj_w, wproj, CC, CC, LDK, 192);
  k_pack_w<<<packGrid((size_t)768 * LDK), 256, 0, stream>>>(fc1_w, wfc1, MLP, CC, LDK, 768);
  k_pack_w<<<packGrid((size_t)192 * MLPP), 256, 0, stream>>>(fc2_w, wfc2, CC, MLP, MLPP, 192);

  // 3) shifted-window gather -> xwin f16
  k_pack_xwin<<<packGrid((size_t)NTOK * LDK), 256, 0, stream>>>(xn, xwin);

  // 4) zero-fill padded operand buffers
  k_fill_h<<<4096, 256, 0, stream>>>(qb, QKN);
  k_fill_h<<<4096, 256, 0, stream>>>(kb, QKN);
  k_fill_h<<<4096, 256, 0, stream>>>(vT, VTN);
  k_fill_h<<<4096, 256, 0, stream>>>(aoutb, (size_t)NTOK * LDK);
  k_fill_h<<<4096, 256, 0, stream>>>(hbuf, (size_t)NTOK * MLPP);

  // 5) QKV GEMM (WMMA, 16x64 strips; 576 = 9 groups of 64 cols)
  k_gemm_qkv<<<dim3(RT, 9), 32, 0, stream>>>(xwin, wq, qkv_b, qb, kb, vT);

  // 6) windowed attention (WMMA + async-to-LDS + softmax)
  k_attn<<<dim3(7, HEADS, NWIN), 32, 0, stream>>>(qb, kb, vT, rpi, rpb, amask, aoutb);

  // 7) proj GEMM (WMMA) + window_reverse/roll scatter -> xa
  k_gemm_proj<<<dim3(RT, 3), 32, 0, stream>>>(aoutb, wproj, proj_b, xa);

  // 8) conv branch
  k_conv1<<<packGrid((size_t)NTOK * 60), 256, 0, stream>>>(xn, conv1_w, conv1_b, t1);
  k_conv2<<<packGrid((size_t)NTOK * CC), 256, 0, stream>>>(t1, conv2_w, conv2_b, ybuf);
  k_pool<<<BB * CC, 256, 0, stream>>>(ybuf, pool);
  k_ca<<<1, 512, 0, stream>>>(pool, ca1_w, ca1_b, ca2_w, ca2_b, asig);

  // 9) residual combine -> d_out
  k_combine<<<packGrid((size_t)NTOK * CC), 256, 0, stream>>>(x, xa, ybuf, asig, out);

  // 10) LN2 -> xm f16 (padded)
  k_ln<<<NTOK / 8, 256, 0, stream>>>(out, g2, b2, nullptr, xm);

  // 11) MLP (WMMA GEMMs, fc2 accumulates residual into d_out)
  k_gemm_fc1<<<dim3(RT, 12), 32, 0, stream>>>(xm, wfc1, fc1_b, hbuf);
  k_gemm_fc2<<<dim3(RT, 3), 32, 0, stream>>>(hbuf, wfc2, fc2_b, out);
}
